// SNNEncoder_43954695307291
// MI455X (gfx1250) — matrix operand order repaired and checked by reading, not compiled, not run on hardware
//
#include <hip/hip_runtime.h>
#include <hip/hip_bf16.h>

// CDNA5 WMMA types
typedef __attribute__((ext_vector_type(16))) _Float16 v16h;
typedef __attribute__((ext_vector_type(8)))  float    v8f;

union Frag16 { uint4 u[2]; v16h h; };

#define LIF_TH 1.0f

__device__ __forceinline__ v8f wmma_f16(v16h a, v16h b, v8f c) {
  // D = A(16x32 f16) * B(32x16 f16) + C(16x16 f32)
  return __builtin_amdgcn_wmma_f32_16x16x32_f16(
      /*neg_a=*/false, a, /*neg_b=*/false, b,
      /*c_mod=*/(short)0, c, /*reuse_a=*/false, /*reuse_b=*/false);
}

// ---------------------------------------------------------------------------
// Layer 0: h = x_t @ W1^T + b1 ; LIF update ; spike ; fr accumulate
// x_t f32 [256][256] staged -> LDS f16 (8x reuse across waves of the block)
// Grid: (4, 16) blocks of 256 threads (8 waves); wave = one 16x16 C tile.
// ---------------------------------------------------------------------------
__global__ __launch_bounds__(256) void snn_layer0(
    const float* __restrict__ x_t,      // [256][256] f32
    const _Float16* __restrict__ Wh,    // [512][256] f16 row-major
    const float* __restrict__ bias,     // [512]
    const float* __restrict__ beta_p,   // scalar
    float* __restrict__ mem,            // [256][512] membrane f32
    _Float16* __restrict__ sh,          // [256][512] spikes f16
    float* __restrict__ fr)             // [256][512] firing-rate accum
{
  __shared__ _Float16 As[16][264];      // +8 pad: 2-way max bank conflict on b128
  const int tid    = threadIdx.x;
  const int m_base = blockIdx.y * 16;

  // cooperative stage + f32->f16 convert: 16 rows x 256 cols, coalesced
  #pragma unroll
  for (int it = 0; it < 16; ++it)
    As[it][tid] = (_Float16)x_t[(m_base + it) * 256 + tid];
  __syncthreads();

  const int wave  = tid >> 5;
  const int lane  = tid & 31;
  const int khalf = lane >> 4;          // 0: K-lo half lanes, 1: K-hi half lanes
  const int arow  = lane & 15;          // A-matrix row M per lane
  const int n_col = blockIdx.x * 128 + wave * 16 + (lane & 15);

  v8f acc = {};
  #pragma unroll
  for (int k0 = 0; k0 < 256; k0 += 32) {
    Frag16 a, b;
    // A 16x32 f16 layout: v0..3 = K[khalf*8 .. +7], v4..7 = K[16+khalf*8 .. +7]
    a.u[0] = *(const uint4*)&As[arow][k0 + khalf * 8];
    a.u[1] = *(const uint4*)&As[arow][k0 + 16 + khalf * 8];
    // B 32x16 f16 layout: v0..3 = K[khalf*8 .. +7], v4..7 = K[16+khalf*8 .. +7] for col n
    const _Float16* bp = Wh + (size_t)n_col * 256 + k0 + khalf * 8;
    b.u[0] = *(const uint4*)bp;
    b.u[1] = *(const uint4*)(bp + 16);
    acc = wmma_f16(a.h, b.h, acc);
  }

  const float be = fminf(fmaxf(beta_p[0], 0.0f), 1.0f);
  const float bv = bias[n_col];
  #pragma unroll
  for (int r = 0; r < 8; ++r) {
    const int    gm  = m_base + r + khalf * 8;   // C layout: M = r + 8*(lane>=16)
    const size_t idx = (size_t)gm * 512 + n_col;
    const float  h   = acc[r] + bv;
    const float  mp  = mem[idx];
    const float  rst = mp > LIF_TH ? 1.0f : 0.0f;
    const float  mn  = be * mp + h - rst * LIF_TH;
    mem[idx] = mn;
    const float s = mn > LIF_TH ? 1.0f : 0.0f;
    sh[idx]  = (_Float16)s;
    fr[idx] += s;
  }
}

// ---------------------------------------------------------------------------
// Layer 1: h2 = s0 @ W2^T + b2 ; LIF ; spike ; fr accumulate. K=512, all f16.
// Spikes are exactly representable in f16 -> GEMM inputs exact.
// ---------------------------------------------------------------------------
__global__ __launch_bounds__(256) void snn_layer1(
    const _Float16* __restrict__ a_in,  // [256][512] f16 spikes
    const _Float16* __restrict__ Wh,    // [512][512] f16
    const float* __restrict__ bias,
    const float* __restrict__ beta_p,
    float* __restrict__ mem,
    _Float16* __restrict__ sh,
    float* __restrict__ fr)
{
  const int tid   = threadIdx.x;
  const int wave  = tid >> 5;
  const int lane  = tid & 31;
  const int khalf = lane >> 4;
  const int m_base = blockIdx.y * 16;
  const int arow   = m_base + (lane & 15);
  const int n_col  = blockIdx.x * 128 + wave * 16 + (lane & 15);

  const _Float16* ap  = a_in + (size_t)arow * 512;
  const _Float16* bp0 = Wh   + (size_t)n_col * 512;

  v8f acc = {};
  #pragma unroll 4
  for (int k0 = 0; k0 < 512; k0 += 32) {
    Frag16 a, b;
    a.u[0] = *(const uint4*)(ap + k0 + khalf * 8);
    a.u[1] = *(const uint4*)(ap + k0 + 16 + khalf * 8);
    const _Float16* bp = bp0 + k0 + khalf * 8;
    b.u[0] = *(const uint4*)bp;
    b.u[1] = *(const uint4*)(bp + 16);
    acc = wmma_f16(a.h, b.h, acc);
  }

  const float be = fminf(fmaxf(beta_p[0], 0.0f), 1.0f);
  const float bv = bias[n_col];
  #pragma unroll
  for (int r = 0; r < 8; ++r) {
    const int    gm  = m_base + r + khalf * 8;
    const size_t idx = (size_t)gm * 512 + n_col;
    const float  h   = acc[r] + bv;
    const float  mp  = mem[idx];
    const float  rst = mp > LIF_TH ? 1.0f : 0.0f;
    const float  mn  = be * mp + h - rst * LIF_TH;
    mem[idx] = mn;
    const float s = mn > LIF_TH ? 1.0f : 0.0f;
    sh[idx]  = (_Float16)s;
    fr[idx] += s;
  }
}

// ---------------------------------------------------------------------------
// Output projection: lat = s1_last @ Wout^T + bout  (256x128, K=512)
// Grid: 16 blocks x 8 waves -> 8 N-tiles cover N=128 per block.
// ---------------------------------------------------------------------------
__global__ __launch_bounds__(256) void final_gemm(
    const _Float16* __restrict__ s1,    // [256][512]
    const _Float16* __restrict__ Wh,    // [128][512]
    const float* __restrict__ bout,     // [128]
    float* __restrict__ lat)            // [256][128]
{
  const int tid   = threadIdx.x;
  const int wave  = tid >> 5;
  const int lane  = tid & 31;
  const int khalf = lane >> 4;
  const int m_base = blockIdx.x * 16;
  const int arow   = m_base + (lane & 15);
  const int n_col  = wave * 16 + (lane & 15);

  const _Float16* ap  = s1 + (size_t)arow * 512;
  const _Float16* bp0 = Wh + (size_t)n_col * 512;

  v8f acc = {};
  #pragma unroll 4
  for (int k0 = 0; k0 < 512; k0 += 32) {
    Frag16 a, b;
    a.u[0] = *(const uint4*)(ap + k0 + khalf * 8);
    a.u[1] = *(const uint4*)(ap + k0 + 16 + khalf * 8);
    const _Float16* bp = bp0 + k0 + khalf * 8;
    b.u[0] = *(const uint4*)bp;
    b.u[1] = *(const uint4*)(bp + 16);
    acc = wmma_f16(a.h, b.h, acc);
  }

  const float bv = bout[n_col];
  #pragma unroll
  for (int r = 0; r < 8; ++r) {
    const int gm = m_base + r + khalf * 8;
    lat[gm * 128 + n_col] = acc[r] + bv;
  }
}

// ---------------------------------------------------------------------------
// LayerNorm over D_lat=128: one block per row
// ---------------------------------------------------------------------------
__global__ __launch_bounds__(128) void ln_kernel(
    const float* __restrict__ lat, const float* __restrict__ gamma,
    const float* __restrict__ beta, float* __restrict__ out)
{
  __shared__ float red[128];
  __shared__ float red2[128];
  const int m = blockIdx.x, t = threadIdx.x;
  const float x = lat[m * 128 + t];
  red[t] = x; red2[t] = x * x;
  __syncthreads();
  #pragma unroll
  for (int s = 64; s > 0; s >>= 1) {
    if (t < s) { red[t] += red[t + s]; red2[t] += red2[t + s]; }
    __syncthreads();
  }
  const float mu  = red[0] * (1.0f / 128.0f);
  const float var = red2[0] * (1.0f / 128.0f) - mu * mu;
  out[m * 128 + t] = (x - mu) * rsqrtf(var + 1e-5f) * gamma[t] + beta[t];
}

// ---------------------------------------------------------------------------
// Utility kernels
// ---------------------------------------------------------------------------
__global__ void f32_to_f16(const float* __restrict__ in, _Float16* __restrict__ out, int n) {
  const int i = blockIdx.x * blockDim.x + threadIdx.x;
  if (i < n) out[i] = (_Float16)in[i];
}

__global__ void zero_f32(float* __restrict__ p, int n) {
  const int i = blockIdx.x * blockDim.x + threadIdx.x;
  if (i < n) p[i] = 0.0f;
}

__global__ void fr_finalize(const float* __restrict__ fr01, float* __restrict__ out,
                            float inv, int n) {
  const int i = blockIdx.x * blockDim.x + threadIdx.x;
  if (i < n) out[i] = fr01[i] * inv;
}

// ---------------------------------------------------------------------------
// Host launch: 1 zero + 3 cvt + 400 GEMM/LIF + gemm + LN + fr (graph-friendly)
// ---------------------------------------------------------------------------
extern "C" void kernel_launch(void* const* d_in, const int* in_sizes, int n_in,
                              void* d_out, int out_size, void* d_ws, size_t ws_size,
                              hipStream_t stream) {
  const float* x    = (const float*)d_in[0];   // [200][256][256]
  const float* W1   = (const float*)d_in[1];   // [512][256]
  const float* b1   = (const float*)d_in[2];
  const float* be0  = (const float*)d_in[3];
  const float* W2   = (const float*)d_in[4];   // [512][512]
  const float* b2   = (const float*)d_in[5];
  const float* be1  = (const float*)d_in[6];
  const float* Wout = (const float*)d_in[7];   // [128][512]
  const float* bout = (const float*)d_in[8];
  const float* lng  = (const float*)d_in[9];
  const float* lnb  = (const float*)d_in[10];
  float* out = (float*)d_out;                  // latent | fr0 | fr1

  // workspace layout (f32 then f16 region) — ~3.5 MB total
  float* ws  = (float*)d_ws;
  float* m0  = ws;                   // 131072 f32
  float* m1  = ws + 131072;          // 131072 f32
  float* fr0 = ws + 262144;          // 131072 f32
  float* fr1 = ws + 393216;          // 131072 f32  (contiguous with fr0)
  _Float16* hb  = (_Float16*)(ws + 524288);
  _Float16* s0h = hb;                // 131072 f16
  _Float16* s1h = hb + 131072;       // 131072 f16
  _Float16* W1h = hb + 262144;       // 131072 f16
  _Float16* W2h = hb + 393216;       // 262144 f16
  _Float16* Woh = hb + 655360;       // 65536  f16
  float* lat = (float*)(hb + 720896);// 32768 f32

  // deterministic per-call init of recurrent state + fr accumulators
  zero_f32<<<2048, 256, 0, stream>>>(m0, 524288);  // m0,m1,fr0,fr1 contiguous

  // weights -> f16 once per call
  f32_to_f16<<<512,  256, 0, stream>>>(W1,   W1h, 131072);
  f32_to_f16<<<1024, 256, 0, stream>>>(W2,   W2h, 262144);
  f32_to_f16<<<256,  256, 0, stream>>>(Wout, Woh, 65536);

  const dim3 grid(4, 16), blk(256);
  for (int t = 0; t < 200; ++t) {
    snn_layer0<<<grid, blk, 0, stream>>>(x + (size_t)t * 65536, W1h, b1, be0, m0, s0h, fr0);
    snn_layer1<<<grid, blk, 0, stream>>>(s0h, W2h, b2, be1, m1, s1h, fr1);
  }

  final_gemm<<<16, 256, 0, stream>>>(s1h, Woh, bout, lat);
  ln_kernel<<<256, 128, 0, stream>>>(lat, lng, lnb, out);
  fr_finalize<<<1024, 256, 0, stream>>>(fr0, out + 32768, 1.0f / 200.0f, 262144);
}
